// MixedLatentAttention_33251636805708
// MI455X (gfx1250) — matrix operand order repaired and checked by reading, not compiled, not run on hardware
//
#include <hip/hip_runtime.h>
#include <hip/hip_bf16.h>
#include <stdint.h>

// ---------------- problem constants ----------------
#define B_   2
#define S_   2048
#define HID_ 2048
#define NH_  16
#define DN_  128
#define DR_  64
#define DV_  128
#define DQK_ 192
#define RANK_ 512
#define M_   (B_ * S_)          // 4096 rows of tokens
#define SCALE_ 0.07216878364870323f   // 192^-0.5

typedef __attribute__((ext_vector_type(16))) __bf16   bf16x16;
typedef __attribute__((ext_vector_type(8)))  float    v8f;
typedef __attribute__((ext_vector_type(8)))  unsigned int u32x8;

static __device__ __forceinline__ unsigned short f2bf(float f) {
  union { float f; unsigned u; } v; v.f = f;
  unsigned r = v.u + 0x7FFFu + ((v.u >> 16) & 1u);
  return (unsigned short)(r >> 16);
}

// A-operand fragment: lane L(<16): row L, K = {k0..k0+7, k0+16..k0+23}
//                     lane L+16  : row L, K = {k0+8..k0+15, k0+24..k0+31}
static __device__ __forceinline__ bf16x16 load_frag_a(const unsigned short* row, int k0, int half) {
  const uint4* p0 = reinterpret_cast<const uint4*>(row + k0 + half * 8);
  const uint4* p1 = reinterpret_cast<const uint4*>(row + k0 + half * 8 + 16);
  uint4 a = *p0, b = *p1;
  u32x8 u = { a.x, a.y, a.z, a.w, b.x, b.y, b.z, b.w };
  return __builtin_bit_cast(bf16x16, u);
}

// B-operand fragment (from Bt = B^T stored [N][K]):
//   lane c(<16): col c, K = {k0..k0+7, k0+16..k0+23} ; lane c+16: col c, K = {k0+8..k0+15, k0+24..k0+31}
static __device__ __forceinline__ bf16x16 load_frag_b(const unsigned short* row, int k0, int half) {
  const uint4* p0 = reinterpret_cast<const uint4*>(row + k0 + half * 8);
  const uint4* p1 = reinterpret_cast<const uint4*>(row + k0 + half * 8 + 16);
  uint4 a = *p0, b = *p1;
  u32x8 u = { a.x, a.y, a.z, a.w, b.x, b.y, b.z, b.w };
  return __builtin_bit_cast(bf16x16, u);
}

static __device__ __forceinline__ v8f wmma_bf(bf16x16 a, bf16x16 b, v8f c) {
  return __builtin_amdgcn_wmma_f32_16x16x32_bf16(false, a, false, b, (short)0, c, false, false);
}

// ---------------- elementwise convert ----------------
__global__ __launch_bounds__(256) void cvt_bf16_kernel(const float* __restrict__ in,
                                                       unsigned short* __restrict__ out, int n) {
  int i = blockIdx.x * 256 + threadIdx.x;
  if (i < n) out[i] = f2bf(in[i]);
}

// W [K][N] f32  ->  Wt [N][K] bf16
__global__ __launch_bounds__(256) void transpose_bf16_kernel(const float* __restrict__ in,
                                                             unsigned short* __restrict__ out,
                                                             int K, int N) {
  int i = blockIdx.x * 256 + threadIdx.x;
  if (i < K * N) {
    int k = i / N, n = i % N;
    out[(size_t)n * K + k] = f2bf(in[i]);
  }
}

// ---------------- generic bf16 WMMA GEMM: C[M][N] = A[M][K] * Bt[N][K]^T ----------------
// grid = (N/64, M/64), block = 32 (one wave); 64x64 tile, 16 WMMAs per K=32 step.
__global__ __launch_bounds__(32) void gemm_bf16_kernel(const unsigned short* __restrict__ A,
                                                       const unsigned short* __restrict__ Bt,
                                                       float* __restrict__ C,
                                                       int Mdim, int Ndim, int Kdim) {
  int lane = threadIdx.x & 31;
  int r    = lane & 15;
  int half = lane >> 4;
  int n0 = blockIdx.x * 64;
  int m0 = blockIdx.y * 64;

  const unsigned short* ap[4];
  const unsigned short* bp[4];
#pragma unroll
  for (int i = 0; i < 4; ++i) {
    ap[i] = A  + (size_t)(m0 + 16 * i + r) * Kdim;
    bp[i] = Bt + (size_t)(n0 + 16 * i + r) * Kdim;
  }

  v8f z = { 0.f, 0.f, 0.f, 0.f, 0.f, 0.f, 0.f, 0.f };
  v8f acc[4][4];
#pragma unroll
  for (int i = 0; i < 4; ++i)
#pragma unroll
    for (int j = 0; j < 4; ++j) acc[i][j] = z;

  for (int k0 = 0; k0 < Kdim; k0 += 32) {
    __builtin_prefetch(ap[0] + k0 + 128, 0, 0);
    __builtin_prefetch(bp[0] + k0 + 128, 0, 0);
    bf16x16 fa[4], fb[4];
#pragma unroll
    for (int i = 0; i < 4; ++i) fa[i] = load_frag_a(ap[i], k0, half);
#pragma unroll
    for (int j = 0; j < 4; ++j) fb[j] = load_frag_b(bp[j], k0, half);
#pragma unroll
    for (int i = 0; i < 4; ++i)
#pragma unroll
      for (int j = 0; j < 4; ++j) acc[i][j] = wmma_bf(fa[i], fb[j], acc[i][j]);
  }

#pragma unroll
  for (int mt = 0; mt < 4; ++mt)
#pragma unroll
    for (int j = 0; j < 4; ++j)
#pragma unroll
      for (int e = 0; e < 8; ++e) {
        int row = m0 + mt * 16 + e + half * 8;
        int col = n0 + j * 16 + r;
        C[(size_t)row * Ndim + col] = acc[mt][j][e];
      }
}

// ---------------- rope on q, reshape to head-major bf16 [B*NH][S][192] ----------------
__global__ __launch_bounds__(256) void rope_q_kernel(const float* __restrict__ q,      // [M][NH*192]
                                                     const float* __restrict__ fc,     // [S][32][2]
                                                     unsigned short* __restrict__ qh) {
  int tid = blockIdx.x * 256 + threadIdx.x;
  if (tid >= M_ * NH_) return;
  int h  = tid % NH_;
  int bs = tid / NH_;
  int s  = bs % S_;
  int b  = bs / S_;
  const float* src = q + (size_t)bs * (NH_ * DQK_) + (size_t)h * DQK_;
  unsigned short* dst = qh + ((size_t)(b * NH_ + h) * S_ + s) * DQK_;
#pragma unroll 4
  for (int i = 0; i < DN_; ++i) dst[i] = f2bf(src[i]);
  const float* f = fc + (size_t)s * 64;
#pragma unroll 4
  for (int p = 0; p < 32; ++p) {
    float c = f[2 * p], sn = f[2 * p + 1];
    float x0 = src[DN_ + 2 * p], x1 = src[DN_ + 2 * p + 1];
    dst[DN_ + 2 * p]     = f2bf(x0 * c - x1 * sn);
    dst[DN_ + 2 * p + 1] = f2bf(x0 * sn + x1 * c);
  }
}

// ---------------- kv_a: layernorm(kv_lat) -> bf16, rope(k_rope) -> f32 ----------------
__global__ __launch_bounds__(256) void kv_prep_kernel(const float* __restrict__ kva,   // [M][576]
                                                      const float* __restrict__ w,
                                                      const float* __restrict__ bias,
                                                      const float* __restrict__ fc,
                                                      unsigned short* __restrict__ kvlat_n, // [M][512] bf16
                                                      float* __restrict__ krope) {          // [M][64]
  int row = blockIdx.x;
  int t   = threadIdx.x;
  const float* x = kva + (size_t)row * (RANK_ + DR_);
  float x0 = x[t], x1 = x[t + 256];
  __shared__ float sred[256];
  sred[t] = x0 + x1;
  __syncthreads();
  for (int s = 128; s > 0; s >>= 1) { if (t < s) sred[t] += sred[t + s]; __syncthreads(); }
  float mu = sred[0] * (1.0f / RANK_);
  __syncthreads();
  float d0 = x0 - mu, d1 = x1 - mu;
  sred[t] = d0 * d0 + d1 * d1;
  __syncthreads();
  for (int s = 128; s > 0; s >>= 1) { if (t < s) sred[t] += sred[t + s]; __syncthreads(); }
  float var  = sred[0] * (1.0f / RANK_);
  float rstd = rsqrtf(var + 1e-5f);
  unsigned short* o = kvlat_n + (size_t)row * RANK_;
  o[t]       = f2bf(d0 * rstd * w[t]       + bias[t]);
  o[t + 256] = f2bf(d1 * rstd * w[t + 256] + bias[t + 256]);
  if (t < 32) {
    int s = row % S_;
    const float* f = fc + (size_t)s * 64;
    float c = f[2 * t], sn = f[2 * t + 1];
    const float* kr = x + RANK_;
    float a0 = kr[2 * t], a1 = kr[2 * t + 1];
    krope[(size_t)row * DR_ + 2 * t]     = a0 * c - a1 * sn;
    krope[(size_t)row * DR_ + 2 * t + 1] = a0 * sn + a1 * c;
  }
}

// ---------------- build K (head-major [B*NH][S][192]) and V^T ([B*NH][128][S]) ----------------
__global__ __launch_bounds__(256) void build_kv_kernel(const float* __restrict__ kv,    // [M][NH*256]
                                                       const float* __restrict__ krope, // [M][64]
                                                       unsigned short* __restrict__ kh,
                                                       unsigned short* __restrict__ vt) {
  int tid = blockIdx.x * 256 + threadIdx.x;
  if (tid >= M_ * NH_) return;
  int h  = tid % NH_;
  int bs = tid / NH_;
  int s  = bs % S_;
  int b  = bs / S_;
  const float* src = kv + (size_t)bs * (NH_ * 256) + (size_t)h * 256;
  unsigned short* kd = kh + ((size_t)(b * NH_ + h) * S_ + s) * DQK_;
#pragma unroll 4
  for (int i = 0; i < DN_; ++i) kd[i] = f2bf(src[i]);
  const float* kr = krope + (size_t)bs * DR_;
#pragma unroll 4
  for (int i = 0; i < DR_; ++i) kd[DN_ + i] = f2bf(kr[i]);
  unsigned short* vd = vt + (size_t)(b * NH_ + h) * DV_ * S_;
#pragma unroll 4
  for (int d = 0; d < DV_; ++d) vd[(size_t)d * S_ + s] = f2bf(src[DN_ + d]);
}

// ---------------- flash attention per head; wave owns 16 q rows ----------------
// grid = (S/64, B*NH), block = 128 (4 waves)
__global__ __launch_bounds__(128) void attn_kernel(const unsigned short* __restrict__ qh_all,
                                                   const unsigned short* __restrict__ kh_all,
                                                   const unsigned short* __restrict__ vt_all,
                                                   unsigned short* __restrict__ attn) { // [M][NH*128] bf16
  int head = blockIdx.y;
  int b    = head / NH_;
  int h    = head % NH_;
  int wave = threadIdx.x >> 5;
  int lane = threadIdx.x & 31;
  int r    = lane & 15;
  int half = lane >> 4;
  int q0   = blockIdx.x * 64 + wave * 16;

  const unsigned short* qh = qh_all + (size_t)head * S_ * DQK_;
  const unsigned short* kh = kh_all + (size_t)head * S_ * DQK_;
  const unsigned short* vt = vt_all + (size_t)head * DV_ * S_;

  __shared__ __align__(128) unsigned short pshared[4][16][32];

  bf16x16 aq[6];
  const unsigned short* qrow = qh + (size_t)(q0 + r) * DQK_;
#pragma unroll
  for (int kk = 0; kk < 6; ++kk) aq[kk] = load_frag_a(qrow, kk * 32, half);

  v8f z = { 0.f, 0.f, 0.f, 0.f, 0.f, 0.f, 0.f, 0.f };
  v8f acc[8];
  float m8[8], l8[8];
#pragma unroll
  for (int i = 0; i < 8; ++i) { acc[i] = z; m8[i] = -1e30f; l8[i] = 0.f; }

  for (int kt = 0; kt <= q0 + 15; kt += 32) {
    v8f s0 = z, s1 = z;
#pragma unroll
    for (int kk = 0; kk < 6; ++kk) {
      bf16x16 b0 = load_frag_b(kh + (size_t)(kt + r) * DQK_, kk * 32, half);
      bf16x16 b1 = load_frag_b(kh + (size_t)(kt + 16 + r) * DQK_, kk * 32, half);
      s0 = wmma_bf(aq[kk], b0, s0);
      s1 = wmma_bf(aq[kk], b1, s1);
    }
    float scalef[8];
#pragma unroll
    for (int e = 0; e < 8; ++e) {
      int rq = q0 + e + half * 8;
      float v0 = s0[e] * SCALE_ + ((kt + r)      > rq ? -1e9f : 0.f);
      float v1 = s1[e] * SCALE_ + ((kt + 16 + r) > rq ? -1e9f : 0.f);
      s0[e] = v0; s1[e] = v1;
      float mx = fmaxf(v0, v1);
      mx = fmaxf(mx, __shfl_xor(mx, 1, 32));
      mx = fmaxf(mx, __shfl_xor(mx, 2, 32));
      mx = fmaxf(mx, __shfl_xor(mx, 4, 32));
      mx = fmaxf(mx, __shfl_xor(mx, 8, 32));
      float mn = fmaxf(m8[e], mx);
      scalef[e] = __expf(m8[e] - mn);
      m8[e] = mn;
    }
#pragma unroll
    for (int e = 0; e < 8; ++e) {
      float p0 = __expf(s0[e] - m8[e]);
      float p1 = __expf(s1[e] - m8[e]);
      int rql = e + half * 8;
      pshared[wave][rql][r]      = f2bf(p0);
      pshared[wave][rql][16 + r] = f2bf(p1);
      float rs = p0 + p1;
      rs += __shfl_xor(rs, 1, 32);
      rs += __shfl_xor(rs, 2, 32);
      rs += __shfl_xor(rs, 4, 32);
      rs += __shfl_xor(rs, 8, 32);
      l8[e] = l8[e] * scalef[e] + rs;
    }
#pragma unroll
    for (int dt = 0; dt < 8; ++dt)
#pragma unroll
      for (int e = 0; e < 8; ++e) acc[dt][e] *= scalef[e];

    asm volatile("s_wait_dscnt 0x0" ::: "memory");  // same-wave LDS RAW fence
    bf16x16 pf = load_frag_a(&pshared[wave][r][0], 0, half);
#pragma unroll
    for (int dt = 0; dt < 8; ++dt) {
      bf16x16 bv = load_frag_b(vt + (size_t)(dt * 16 + r) * S_, kt, half);
      acc[dt] = wmma_bf(pf, bv, acc[dt]);
    }
  }

  float inv[8];
#pragma unroll
  for (int e = 0; e < 8; ++e) inv[e] = 1.0f / l8[e];
#pragma unroll
  for (int dt = 0; dt < 8; ++dt)
#pragma unroll
    for (int e = 0; e < 8; ++e) {
      int sq  = q0 + e + half * 8;
      int col = h * DV_ + dt * 16 + r;
      attn[(size_t)(b * S_ + sq) * (NH_ * DV_) + col] = f2bf(acc[dt][e] * inv[e]);
    }
}

// ---------------- workspace layout (bytes) ----------------
static constexpr size_t SZ_HID_BF = (size_t)M_ * HID_ * 2;            // 16.8 MB
static constexpr size_t SZ_WQT    = (size_t)(NH_ * DQK_) * HID_ * 2;  // 12.6 MB
static constexpr size_t SZ_WKVAT  = (size_t)(RANK_ + DR_) * HID_ * 2; // 2.36 MB
static constexpr size_t SZ_QF     = (size_t)M_ * (NH_ * DQK_) * 4;    // 50.3 MB
static constexpr size_t SZ_KVF    = (size_t)M_ * (NH_ * 256) * 4;     // 67.1 MB
static constexpr size_t O_HID   = 0;
static constexpr size_t O_WQT   = O_HID + SZ_HID_BF;
static constexpr size_t O_WKVAT = O_WQT + SZ_WQT;
static constexpr size_t O_QF    = O_WKVAT + SZ_WKVAT;
static constexpr size_t O_KVF   = 0;                                   // overlays dead phase-A region
static constexpr size_t REG0    = (O_QF + SZ_QF > SZ_KVF) ? (O_QF + SZ_QF) : SZ_KVF;
static constexpr size_t O_KVA   = REG0;                                // f32 [M][576]
static constexpr size_t SZ_KVA  = (size_t)M_ * (RANK_ + DR_) * 4;
static constexpr size_t O_KVLAT = O_KVA + SZ_KVA;                      // bf16 [M][512]
static constexpr size_t SZ_KVLAT= (size_t)M_ * RANK_ * 2;
static constexpr size_t O_KROPE = O_KVLAT + SZ_KVLAT;                  // f32 [M][64]
static constexpr size_t SZ_KROPE= (size_t)M_ * DR_ * 4;
static constexpr size_t O_WKVBT = O_KROPE + SZ_KROPE;                  // bf16 [4096][512]
static constexpr size_t SZ_WKVBT= (size_t)(NH_ * 256) * RANK_ * 2;
static constexpr size_t O_WOT   = O_WKVBT + SZ_WKVBT;                  // bf16 [2048][2048]
static constexpr size_t SZ_WOT  = (size_t)HID_ * (NH_ * DV_) * 2;
static constexpr size_t O_QH    = O_WOT + SZ_WOT;                      // bf16 [B*NH][S][192]
static constexpr size_t SZ_QH   = (size_t)B_ * NH_ * S_ * DQK_ * 2;
static constexpr size_t O_KH    = O_QH + SZ_QH;
static constexpr size_t SZ_KH   = SZ_QH;
static constexpr size_t O_VT    = O_KH + SZ_KH;                        // bf16 [B*NH][128][S]
static constexpr size_t SZ_VT   = (size_t)B_ * NH_ * DV_ * S_ * 2;
static constexpr size_t O_ATTN  = O_VT + SZ_VT;                        // bf16 [M][2048]

extern "C" void kernel_launch(void* const* d_in, const int* in_sizes, int n_in,
                              void* d_out, int out_size, void* d_ws, size_t ws_size,
                              hipStream_t stream) {
  const float* hidden = (const float*)d_in[0];
  const float* fc     = (const float*)d_in[1];
  // d_in[2] = attention_mask (pure causal; computed in-kernel)
  const float* Wq     = (const float*)d_in[3];
  const float* Wkv_a  = (const float*)d_in[4];
  const float* kvw    = (const float*)d_in[5];
  const float* kvb    = (const float*)d_in[6];
  const float* Wkv_b  = (const float*)d_in[7];
  const float* Wo     = (const float*)d_in[8];
  float* out = (float*)d_out;

  char* ws = (char*)d_ws;
  unsigned short* hid_bf  = (unsigned short*)(ws + O_HID);
  unsigned short* WqT     = (unsigned short*)(ws + O_WQT);
  unsigned short* WkvaT   = (unsigned short*)(ws + O_WKVAT);
  float*          q_f32   = (float*)(ws + O_QF);
  float*          kv_f32  = (float*)(ws + O_KVF);
  float*          kva_f32 = (float*)(ws + O_KVA);
  unsigned short* kvlat_n = (unsigned short*)(ws + O_KVLAT);
  float*          krope   = (float*)(ws + O_KROPE);
  unsigned short* WkvbT   = (unsigned short*)(ws + O_WKVBT);
  unsigned short* WoT     = (unsigned short*)(ws + O_WOT);
  unsigned short* qh      = (unsigned short*)(ws + O_QH);
  unsigned short* kh      = (unsigned short*)(ws + O_KH);
  unsigned short* vt      = (unsigned short*)(ws + O_VT);
  unsigned short* attn_bf = (unsigned short*)(ws + O_ATTN);

  auto cdiv = [](int a, int b) { return (a + b - 1) / b; };

  // phase A: precision conversion + weight transposes
  cvt_bf16_kernel<<<cdiv(M_ * HID_, 256), 256, 0, stream>>>(hidden, hid_bf, M_ * HID_);
  transpose_bf16_kernel<<<cdiv(HID_ * NH_ * DQK_, 256), 256, 0, stream>>>(Wq, WqT, HID_, NH_ * DQK_);
  transpose_bf16_kernel<<<cdiv(HID_ * (RANK_ + DR_), 256), 256, 0, stream>>>(Wkv_a, WkvaT, HID_, RANK_ + DR_);
  transpose_bf16_kernel<<<cdiv(RANK_ * NH_ * 256, 256), 256, 0, stream>>>(Wkv_b, WkvbT, RANK_, NH_ * 256);
  transpose_bf16_kernel<<<cdiv(NH_ * DV_ * HID_, 256), 256, 0, stream>>>(Wo, WoT, NH_ * DV_, HID_);

  // q = hidden @ Wq ; kv_a = hidden @ Wkv_a
  gemm_bf16_kernel<<<dim3((NH_ * DQK_) / 64, M_ / 64), 32, 0, stream>>>(hid_bf, WqT, q_f32, M_, NH_ * DQK_, HID_);
  gemm_bf16_kernel<<<dim3((RANK_ + DR_) / 64, M_ / 64), 32, 0, stream>>>(hid_bf, WkvaT, kva_f32, M_, RANK_ + DR_, HID_);

  // phase B: rope q, layernorm+rope kv_a   (q_f32 & hid_bf dead afterwards)
  rope_q_kernel<<<cdiv(M_ * NH_, 256), 256, 0, stream>>>(q_f32, fc, qh);
  kv_prep_kernel<<<M_, 256, 0, stream>>>(kva_f32, kvw, kvb, fc, kvlat_n, krope);

  // phase C: kv = ln(kv_lat) @ Wkv_b   (kv_f32 overlays phase-A region)
  gemm_bf16_kernel<<<dim3((NH_ * 256) / 64, M_ / 64), 32, 0, stream>>>(kvlat_n, WkvbT, kv_f32, M_, NH_ * 256, RANK_);

  // phase D: head-major K and transposed V
  build_kv_kernel<<<cdiv(M_ * NH_, 256), 256, 0, stream>>>(kv_f32, krope, kh, vt);

  // phase E: flash attention
  attn_kernel<<<dim3(S_ / 64, B_ * NH_), 128, 0, stream>>>(qh, kh, vt, attn_bf);

  // phase F: out = attn @ Wo  -> d_out (f32)
  gemm_bf16_kernel<<<dim3(HID_ / 64, M_ / 64), 32, 0, stream>>>(attn_bf, WoT, out, M_, HID_, NH_ * DV_);
}
